// MultiHeadAttentionLayer_19310172963271
// MI455X (gfx1250) — matrix operand order repaired and checked by reading, not compiled, not run on hardware
//
#include <hip/hip_runtime.h>

// ---------------- MI455X / gfx1250 fused MHA ----------------
// q,k,v [B,S,D] f32 -> (QKV proj f16) -> flash attention (WMMA f16, f32 acc,
// V tile staged via Tensor Data Mover when available) -> output projection.
// All matrix math on v_wmma_f32_16x16x32_f16.

constexpr int B_ = 4;
constexpr int S_ = 2048;
constexpr int D_ = 1024;
constexpr int H_ = 16;
constexpr int HD_ = 64;
constexpr int M_ = B_ * S_;   // 8192 tokens

typedef _Float16 v16h __attribute__((ext_vector_type(16)));
typedef _Float16 v8h  __attribute__((ext_vector_type(8)));
typedef float    v8f  __attribute__((ext_vector_type(8)));
typedef float    v4f  __attribute__((ext_vector_type(4)));
typedef unsigned int u32x4 __attribute__((ext_vector_type(4)));
typedef int      i32x4 __attribute__((ext_vector_type(4)));
typedef int      i32x8 __attribute__((ext_vector_type(8)));

union V16U { v16h v; v8h h[2]; };

#if defined(__has_builtin)
#if __has_builtin(__builtin_amdgcn_tensor_load_to_lds) && \
    __has_builtin(__builtin_amdgcn_s_wait_tensorcnt)
#define HAVE_TDM 1
#endif
#endif
#ifndef HAVE_TDM
#define HAVE_TDM 0
#endif

__device__ __forceinline__ v8f wmma_f16(v16h a, v16h b, v8f c) {
  // D = A(16x32 f16) * B(32x16 f16) + C(16x16 f32)
  return __builtin_amdgcn_wmma_f32_16x16x32_f16(
      /*neg_a=*/false, a, /*neg_b=*/false, b,
      /*c_mod=*/(short)0, c, /*reuse_a=*/false, /*reuse_b=*/false);
}

#if HAVE_TDM
// Issue one TDM 2-D tile load: tile_h rows x tile_w f16 elems, row stride
// row_stride elems, from gaddr into LDS byte offset lds_byte_off.
__device__ __forceinline__ void tdm_load_2d_f16(
    unsigned int lds_byte_off, const void* gaddr,
    unsigned int tensor_w, unsigned int tensor_h,
    unsigned int tile_w, unsigned int tile_h, unsigned int row_stride)
{
  const unsigned long long ga = (unsigned long long)(uintptr_t)gaddr;
  u32x4 g0;
  g0[0] = 1u;                                       // count=1, user mode
  g0[1] = lds_byte_off;                             // D#.lds_addr
  g0[2] = (unsigned int)(ga & 0xFFFFFFFFu);         // global_addr[31:0]
  g0[3] = (unsigned int)((ga >> 32) & 0x01FFFFFFu)  // global_addr[56:32]
        | (2u << 30);                               // type = 2 (image)
  i32x8 g1;
  g1[0] = (int)(1u << 16);                          // data_size=1 (2 bytes)
  g1[1] = (int)((tensor_w & 0xFFFFu) << 16);        // tensor_dim0[15:0]
  g1[2] = (int)(((tensor_w >> 16) & 0xFFFFu) |      // tensor_dim0[31:16]
                ((tensor_h & 0xFFFFu) << 16));      // tensor_dim1[15:0]
  g1[3] = (int)(((tensor_h >> 16) & 0xFFFFu) |      // tensor_dim1[31:16]
                ((tile_w & 0xFFFFu) << 16));        // tile_dim0
  g1[4] = (int)(tile_h & 0xFFFFu);                  // tile_dim1 (tile_dim2=0)
  g1[5] = (int)row_stride;                          // tensor_dim0_stride[31:0]
  g1[6] = 0;                                        // stride hi / dim1_stride
  g1[7] = 0;
  const i32x4 z4 = {0, 0, 0, 0};
#if __clang_major__ >= 23
  const i32x8 z8 = {0, 0, 0, 0, 0, 0, 0, 0};
  __builtin_amdgcn_tensor_load_to_lds(g0, g1, z4, z4, z8, 0);
#else
  __builtin_amdgcn_tensor_load_to_lds(g0, g1, z4, z4, 0);
#endif
}
#endif // HAVE_TDM

// ---------------------------------------------------------------------------
// GEMM: C[M,N] = A[M,K] * W[N,K]^T + bias[N]
// block = 256 threads (8 waves), tile 64x64, BK=32.
// wave w: rows m0=(w&3)*16, cols n0=(w>>2)*32 -> two 16x16 accumulators.
// ---------------------------------------------------------------------------
template <typename AT, typename OT>
__global__ __launch_bounds__(256) void gemm_bias_wmma(
    const AT* __restrict__ A, const float* __restrict__ W,
    const float* __restrict__ bias, OT* __restrict__ Cmat,
    int M, int N, int K)
{
  constexpr int BM = 64, BN = 64, BK = 32, LDT = 40; // 40-half stride: 80B, 16B aligned
  __shared__ _Float16 As[BM][LDT];
  __shared__ _Float16 Bs[BN][LDT];

  const int tid  = threadIdx.x;
  const int lane = tid & 31;
  const int wave = tid >> 5;
  const int mBase = blockIdx.y * BM;
  const int nBase = blockIdx.x * BN;
  const int m0 = (wave & 3) * 16;
  const int n0 = (wave >> 2) * 32;

  const int lr = tid >> 2;          // 0..63 tile row loaded by this thread
  const int lc = (tid & 3) * 8;     // 0,8,16,24 k-chunk

  const int hi = lane >> 4;         // half-wave select
  const int nl = lane & 15;
  const int kbA = hi * 8;           // unused; A-frag also takes K = kbB + i
  const int kbB = hi * 16;          // B-frag: K = kbB + i

  v8f acc[2] = {};

  for (int k0 = 0; k0 < K; k0 += BK) {
    // ---- stage A tile (convert f32->f16 if needed) ----
    const AT* asrc = A + (size_t)(mBase + lr) * K + (k0 + lc);
    if constexpr (sizeof(AT) == 4) {
      v4f f0 = *(const v4f*)(asrc);
      v4f f1 = *(const v4f*)(asrc + 4);
      #pragma unroll
      for (int i = 0; i < 4; ++i) {
        As[lr][lc + i]     = (_Float16)f0[i];
        As[lr][lc + 4 + i] = (_Float16)f1[i];
      }
    } else {
      *(v8h*)&As[lr][lc] = *(const v8h*)(asrc);
    }
    // ---- stage W tile (always f32 -> f16) as Bs[n][k] ----
    const float* wsrc = W + (size_t)(nBase + lr) * K + (k0 + lc);
    v4f g0 = *(const v4f*)(wsrc);
    v4f g1 = *(const v4f*)(wsrc + 4);
    #pragma unroll
    for (int i = 0; i < 4; ++i) {
      Bs[lr][lc + i]     = (_Float16)g0[i];
      Bs[lr][lc + 4 + i] = (_Float16)g1[i];
    }
    // prefetch next tiles while this one computes
    if (k0 + BK < K) {
      __builtin_prefetch(asrc + BK, 0, 1);
      __builtin_prefetch(wsrc + BK, 0, 1);
    }
    __syncthreads();

    // ---- fragments from LDS ----
    V16U af;
    const int ar = m0 + nl;
    af.h[0] = *(const v8h*)&As[ar][kbB];
    af.h[1] = *(const v8h*)&As[ar][kbB + 8];

    #pragma unroll
    for (int t = 0; t < 2; ++t) {
      V16U bf;
      const int bc = n0 + t * 16 + nl;
      bf.h[0] = *(const v8h*)&Bs[bc][kbB];
      bf.h[1] = *(const v8h*)&Bs[bc][kbB + 8];
      acc[t] = wmma_f16(af.v, bf.v, acc[t]);
    }
    __syncthreads();
  }

  // ---- epilogue: bias + store (C layout: vgpr r -> M = r + 8*hi, N = nl) ----
  #pragma unroll
  for (int t = 0; t < 2; ++t) {
    const int gn = nBase + n0 + t * 16 + nl;
    const float bv = bias[gn];
    #pragma unroll
    for (int r = 0; r < 8; ++r) {
      const int gm = mBase + m0 + r + hi * 8;
      Cmat[(size_t)gm * N + gn] = (OT)(acc[t][r] + bv);
    }
  }
}

// ---------------------------------------------------------------------------
// Flash attention (causal). grid = (S/64, H, B), block = 128 (4 waves).
// Each wave owns 16 query rows; key blocks of 32. V tile staged to per-wave
// LDS by the Tensor Data Mover (TENSORcnt), P converted C->A layout via LDS.
// Layout of Qh/Kh/Vh: [B,S,H,HD] f16.
// ---------------------------------------------------------------------------
__global__ __launch_bounds__(128) void flash_attn_wmma(
    const _Float16* __restrict__ Qh, const _Float16* __restrict__ Kh,
    const _Float16* __restrict__ Vh, _Float16* __restrict__ Oh)
{
  constexpr int LDP = 40;
  __shared__ _Float16 Pl[4][16][LDP];
#if HAVE_TDM
  __shared__ _Float16 Vt[4][32][HD_];   // per-wave V tile (32 keys x 64 dims)
#endif

  const int lane = threadIdx.x & 31;
  const int wave = threadIdx.x >> 5;
  const int b = blockIdx.z;
  const int h = blockIdx.y;
  const int q0 = blockIdx.x * 64 + wave * 16;
  const int hi = lane >> 4;
  const int nl = lane & 15;
  const int kbA = hi * 8;
  const int kbB = hi * 16;
  const float scale = 0.125f; // 1/sqrt(64)

  // ---- Q A-fragments: rows q0..q0+15, head-dim chunks {0..31},{32..63} ----
  const _Float16* qrow = Qh + ((size_t)b * S_ + (q0 + nl)) * D_ + h * HD_;
  V16U qf[2];
  #pragma unroll
  for (int c = 0; c < 2; ++c) {
    qf[c].h[0] = *(const v8h*)(qrow + c * 32 + kbB);
    qf[c].h[1] = *(const v8h*)(qrow + c * 32 + kbB + 8);
  }

  v8f oacc[4] = {};
  float mrun[8], lrun[8];
  #pragma unroll
  for (int r = 0; r < 8; ++r) { mrun[r] = -1e30f; lrun[r] = 0.0f; }

  const _Float16* kbase = Kh + (size_t)b * S_ * D_ + h * HD_;
  const _Float16* vbase = Vh + (size_t)b * S_ * D_ + h * HD_;

#if HAVE_TDM
  const unsigned int vt_off = (unsigned int)(uintptr_t)
      ((__attribute__((address_space(3))) void*)&Vt[wave][0][0]);
#endif

  const int kend = q0 + 16;                 // causal: keys <= q0+15
  for (int j0 = 0; j0 < kend; j0 += 32) {
#if HAVE_TDM
    // ---- kick off async V-tile DMA (overlaps with score WMMAs) ----
    tdm_load_2d_f16(vt_off, vbase + (size_t)j0 * D_,
                    /*tensor_w=*/HD_, /*tensor_h=*/(unsigned)(S_ - j0),
                    /*tile_w=*/HD_, /*tile_h=*/32, /*row_stride=*/D_);
#endif
    if (j0 + 32 < kend) {
      __builtin_prefetch(kbase + (size_t)(j0 + 32 + nl) * D_, 0, 1);
    }
    // ---- S = Q * K^T : two 16x16 n-tiles, K-dim=64 in two WMMA each ----
    v8f sacc[2];
    #pragma unroll
    for (int t = 0; t < 2; ++t) {
      v8f a = {};
      const _Float16* krow = kbase + (size_t)(j0 + t * 16 + nl) * D_;
      #pragma unroll
      for (int c = 0; c < 2; ++c) {
        V16U kf;
        kf.h[0] = *(const v8h*)(krow + c * 32 + kbB);
        kf.h[1] = *(const v8h*)(krow + c * 32 + kbB + 8);
        a = wmma_f16(qf[c].v, kf.v, a);
      }
      sacc[t] = a;
    }
    // ---- causal mask + online softmax (row stats per accumulator vgpr) ----
    #pragma unroll
    for (int r = 0; r < 8; ++r) {
      const int gm = q0 + r + hi * 8;
      float s0 = sacc[0][r] * scale;
      float s1 = sacc[1][r] * scale;
      if (j0 + nl > gm)      s0 = -1e30f;
      if (j0 + 16 + nl > gm) s1 = -1e30f;
      float mx = fmaxf(s0, s1);
      #pragma unroll
      for (int off = 8; off >= 1; off >>= 1)
        mx = fmaxf(mx, __shfl_xor(mx, off, 32));
      const float mnew  = fmaxf(mrun[r], mx);
      const float alpha = __expf(mrun[r] - mnew);
      const float p0 = __expf(s0 - mnew);
      const float p1 = __expf(s1 - mnew);
      float ps = p0 + p1;
      #pragma unroll
      for (int off = 8; off >= 1; off >>= 1)
        ps += __shfl_xor(ps, off, 32);
      lrun[r] = lrun[r] * alpha + ps;
      mrun[r] = mnew;
      #pragma unroll
      for (int d = 0; d < 4; ++d) oacc[d][r] *= alpha;
      // stash P in per-wave LDS for C-layout -> A-layout conversion
      Pl[wave][r + hi * 8][nl]      = (_Float16)p0;
      Pl[wave][r + hi * 8][16 + nl] = (_Float16)p1;
    }
    asm volatile("s_wait_dscnt 0" ::: "memory"); // in-wave LDS RAW
    // ---- P as A-fragment (16x32) ----
    V16U pf;
    pf.h[0] = *(const v8h*)&Pl[wave][nl][kbB];
    pf.h[1] = *(const v8h*)&Pl[wave][nl][kbB + 8];
    // ---- O += P * V : 4 n-tiles over head dim ----
#if HAVE_TDM
    __builtin_amdgcn_s_wait_tensorcnt(0);        // V tile resident in LDS
    asm volatile("" ::: "memory");
    #pragma unroll
    for (int d = 0; d < 4; ++d) {
      v16h vf;
      #pragma unroll
      for (int i = 0; i < 16; ++i)
        vf[i] = Vt[wave][kbB + i][d * 16 + nl];
      oacc[d] = wmma_f16(pf.v, vf, oacc[d]);
    }
#else
    #pragma unroll
    for (int d = 0; d < 4; ++d) {
      v16h vf;
      #pragma unroll
      for (int i = 0; i < 16; ++i)
        vf[i] = vbase[(size_t)(j0 + kbB + i) * D_ + d * 16 + nl];
      oacc[d] = wmma_f16(pf.v, vf, oacc[d]);
    }
#endif
  }

  // ---- normalize and store f16 ----
  _Float16* orow = Oh + ((size_t)b * S_ + q0) * D_ + h * HD_;
  #pragma unroll
  for (int r = 0; r < 8; ++r) {
    const float inv = 1.0f / lrun[r];
    const int m = r + hi * 8;
    #pragma unroll
    for (int d = 0; d < 4; ++d)
      orow[(size_t)m * D_ + d * 16 + nl] = (_Float16)(oacc[d][r] * inv);
  }
}

// ---------------------------------------------------------------------------
extern "C" void kernel_launch(void* const* d_in, const int* in_sizes, int n_in,
                              void* d_out, int out_size, void* d_ws, size_t ws_size,
                              hipStream_t stream)
{
  (void)in_sizes; (void)n_in; (void)out_size; (void)ws_size;
  // setup_inputs order: q k v mask Wq bq Wk bk Wv bv Wo bo
  const float* q  = (const float*)d_in[0];
  const float* k  = (const float*)d_in[1];
  const float* v  = (const float*)d_in[2];
  const float* Wq = (const float*)d_in[4];
  const float* bq = (const float*)d_in[5];
  const float* Wk = (const float*)d_in[6];
  const float* bk = (const float*)d_in[7];
  const float* Wv = (const float*)d_in[8];
  const float* bv = (const float*)d_in[9];
  const float* Wo = (const float*)d_in[10];
  const float* bo = (const float*)d_in[11];
  float* out = (float*)d_out;

  const size_t tok = (size_t)M_ * D_;          // 8M halves = 16 MB each
  _Float16* qh = (_Float16*)d_ws;
  _Float16* kh = qh + tok;
  _Float16* vh = kh + tok;
  _Float16* ah = vh + tok;

  dim3 gg(D_ / 64, M_ / 64);                   // (16, 128)
  gemm_bias_wmma<float, _Float16><<<gg, 256, 0, stream>>>(q, Wq, bq, qh, M_, D_, D_);
  gemm_bias_wmma<float, _Float16><<<gg, 256, 0, stream>>>(k, Wk, bk, kh, M_, D_, D_);
  gemm_bias_wmma<float, _Float16><<<gg, 256, 0, stream>>>(v, Wv, bv, vh, M_, D_, D_);

  dim3 ga(S_ / 64, H_, B_);                    // (32, 16, 4)
  flash_attn_wmma<<<ga, 128, 0, stream>>>(qh, kh, vh, ah);

  gemm_bias_wmma<_Float16, float><<<gg, 256, 0, stream>>>(ah, Wo, bo, out, M_, D_, D_);
}
